// ROIAwareMP_81767587381703
// MI455X (gfx1250) — matrix operand order repaired and checked by reading, not compiled, not run on hardware
//
#include <hip/hip_runtime.h>

// ---------------------------------------------------------------------------
// PointNetConv x2 on MI455X (gfx1250), wave32 + WMMA f16 + async LDS staging.
//
// Per layer:
//   edge kernel : persistent waves, 16 edges/tile. msg(16x256 f16 in LDS)
//                 x lw1T -> ReLU -> (16x128 f16 in LDS) x lw2T -> atomic
//                 segment-max into agg. v_wmma_f32_16x16x32_f16 throughout.
//   node kernel : persistent waves, 16 nodes/tile. agg x gw1T -> ReLU -> gw2T.
//
// Weights staged once per block via GLOBAL_LOAD_ASYNC_TO_LDS_B128 (ASYNCcnt),
// falling back to int4 copies if the builtin is unavailable.
//
// FLOP total ~2.2e11 (f16 WMMA); gather traffic ~3.4 GB (~150us @ 23.3TB/s);
// 2.85e8 atomics dominate in practice (inherent to unsorted segment_max).
// ---------------------------------------------------------------------------

#define NNODES 65536
#define NEDGES 1048576
#define HID    128
#define ETOT   (NEDGES + NNODES)   // 1114112, divisible by 16

#define EDGE_BLOCKS 2048
#define NODE_BLOCKS 256

typedef __attribute__((ext_vector_type(16))) _Float16 v16h;
typedef __attribute__((ext_vector_type(8)))  _Float16 v8h;
typedef __attribute__((ext_vector_type(8)))  float    v8f;
typedef __attribute__((ext_vector_type(4)))  int      v4i;

#if defined(__has_builtin)
#if __has_builtin(__builtin_amdgcn_global_load_async_to_lds_b128)
#define HAVE_ASYNC_LDS 1
#endif
#endif

#if defined(HAVE_ASYNC_LDS)
typedef __attribute__((address_space(1))) v4i* gv4i_t;   // global int4*
typedef __attribute__((address_space(3))) v4i* lv4i_t;   // LDS int4*
#endif

// Stage `bytes` (multiple of 16) from global to LDS, cooperatively.
__device__ __forceinline__ void stage_lds(const void* g, void* s, int bytes,
                                          int tid, int nthreads) {
#if defined(HAVE_ASYNC_LDS)
    gv4i_t gp = (gv4i_t)g;
    lv4i_t sp = (lv4i_t)s;
    for (int i = tid; i < bytes / 16; i += nthreads)
        __builtin_amdgcn_global_load_async_to_lds_b128(gp + i, sp + i, 0, 0);
#else
    const int4* gp = (const int4*)g;
    int4*       sp = (int4*)s;
    for (int i = tid; i < bytes / 16; i += nthreads) sp[i] = gp[i];
#endif
}

__device__ __forceinline__ void stage_fence() {
#if defined(HAVE_ASYNC_LDS)
    asm volatile("s_wait_asynccnt 0x0" ::: "memory");
#endif
    __syncthreads();
}

// Monotonic int mapping: exact float max via integer atomics.
__device__ __forceinline__ void atomic_max_f32(float* addr, float val) {
    if (!(val < 0.0f)) atomicMax((int*)addr, __float_as_int(val));
    else               atomicMin((unsigned int*)addr, __float_as_uint(val));
}

// A-fragment (16-bit A 16x32 layout): lane holds row m; elements 0..15 at
// K = kb*32 + (lane>=16 ? 16 : 0) + s (contiguous).
__device__ __forceinline__ v16h load_a_frag(const _Float16* row, int kb, int lane) {
    const int k0 = kb * 32 + ((lane & 16) ? 16 : 0);
    v8h c0 = *(const v8h*)(row + k0);
    v8h c1 = *(const v8h*)(row + k0 + 8);
    v16h a;
#pragma unroll
    for (int j = 0; j < 8; ++j) { a[j] = c0[j]; a[8 + j] = c1[j]; }
    return a;
}

__device__ __forceinline__ v8h cvt8(float4 p0, float4 p1) {
    v8h h;
    h[0] = (_Float16)p0.x; h[1] = (_Float16)p0.y; h[2] = (_Float16)p0.z; h[3] = (_Float16)p0.w;
    h[4] = (_Float16)p1.x; h[5] = (_Float16)p1.y; h[6] = (_Float16)p1.z; h[7] = (_Float16)p1.w;
    return h;
}

// Convert W (din x HID f32, row-major) -> WT (HID x din f16): WT[n][k] = W[k][n]
__global__ void k_w2h(const float* __restrict__ W, _Float16* __restrict__ WT, int din) {
    int i = blockIdx.x * blockDim.x + threadIdx.x;
    if (i >= din * HID) return;
    int n = i & (HID - 1), k = i >> 7;
    WT[n * din + k] = (_Float16)W[k * HID + n];
}

__global__ void k_fill(float* __restrict__ p, float v, int n) {
    int i = blockIdx.x * blockDim.x + threadIdx.x;
    if (i < n) p[i] = v;
}

// ---------------------------------------------------------------------------
// Edge kernel: 4 waves/block, persistent; 16 edges per tile iteration.
// ---------------------------------------------------------------------------
__global__ __launch_bounds__(128) void k_edge(
    const float* __restrict__ xin, const float* __restrict__ pos,
    const long long* __restrict__ ei,
    const _Float16* __restrict__ lw1T,  // 128 x 256 (f16, transposed)
    const _Float16* __restrict__ lw2T,  // 128 x 128
    const float* __restrict__ lb1, const float* __restrict__ lb2,
    float* __restrict__ agg)
{
    extern __shared__ char smem[];
    _Float16* sW1      = (_Float16*)smem;            // 128*256 halves
    _Float16* sW2      = sW1 + 128 * 256;            // 128*128
    _Float16* sMsgBase = sW2 + 128 * 128;            // 4 waves * 16*256
    _Float16* sHidBase = sMsgBase + 4 * (16 * 256);  // 4 waves * 16*128
    int*      sDstBase = (int*)(sHidBase + 4 * (16 * 128));

    const int tid  = threadIdx.x;
    const int lane = tid & 31;
    const int wave = tid >> 5;
    _Float16* sMsg = sMsgBase + wave * (16 * 256);
    _Float16* sHid = sHidBase + wave * (16 * 128);
    int*      sDst = sDstBase + wave * 16;

    // Stage weights once per block (async direct-to-LDS).
    stage_lds(lw1T, sW1, 128 * 256 * 2, tid, blockDim.x);
    stage_lds(lw2T, sW2, 128 * 128 * 2, tid, blockDim.x);
    stage_fence();

    const int m    = lane & 15;
    const int ncol = lane & 15;
    const int bofs = (lane & 16) ? 16 : 0;   // B frag K base within K-block
    const int mofs = (lane & 16) ? 8 : 0;    // C frag row base
    const int NT   = ETOT / 16;

    for (int tile = blockIdx.x * 4 + wave; tile < NT; tile += EDGE_BLOCKS * 4) {
        const int e = tile * 16 + m;
        int s, d;
        if (e < NEDGES) { s = (int)ei[e]; d = (int)ei[NEDGES + e]; }
        else            { s = e - NEDGES; d = s; }
        if (lane < 16) sDst[m] = d;

        // Build 16x256 f16 msg tile: lanes 0-15 -> x[src], lanes 16-31 -> pos diff.
        if (lane < 16) {
            const float* xr = xin + (size_t)s * HID;
#pragma unroll
            for (int k = 0; k < HID; k += 8) {
                float4 p0 = *(const float4*)(xr + k);
                float4 p1 = *(const float4*)(xr + k + 4);
                *(v8h*)(sMsg + m * 256 + k) = cvt8(p0, p1);
            }
        } else {
            const float* ps = pos + (size_t)s * HID;
            const float* pd = pos + (size_t)d * HID;
#pragma unroll
            for (int k = 0; k < HID; k += 8) {
                float4 a0 = *(const float4*)(ps + k);
                float4 a1 = *(const float4*)(ps + k + 4);
                float4 b0 = *(const float4*)(pd + k);
                float4 b1 = *(const float4*)(pd + k + 4);
                float4 q0 = make_float4(a0.x - b0.x, a0.y - b0.y, a0.z - b0.z, a0.w - b0.w);
                float4 q1 = make_float4(a1.x - b1.x, a1.y - b1.y, a1.z - b1.z, a1.w - b1.w);
                *(v8h*)(sMsg + m * 256 + 128 + k) = cvt8(q0, q1);
            }
        }
        // Tile is wave-private: only this wave's DS writes must land.
        asm volatile("s_wait_dscnt 0x0" ::: "memory");

        // GEMM1: (16x256) x (256x128), K in 8 blocks of 32.
        v16h afr[8];
#pragma unroll
        for (int kb = 0; kb < 8; ++kb) afr[kb] = load_a_frag(sMsg + m * 256, kb, lane);

#pragma unroll
        for (int nt = 0; nt < 8; ++nt) {
            const int n = nt * 16 + ncol;
            v8f acc = {};
#pragma unroll
            for (int kb = 0; kb < 8; ++kb) {
                const v16h b = *(const v16h*)(sW1 + n * 256 + kb * 32 + bofs);
                acc = __builtin_amdgcn_wmma_f32_16x16x32_f16(false, afr[kb], false, b,
                                                             (short)0, acc, false, false);
            }
            const float bias = lb1[n];
#pragma unroll
            for (int i = 0; i < 8; ++i) {
                float v = acc[i] + bias;
                v = v > 0.0f ? v : 0.0f;
                sHid[(mofs + i) * 128 + n] = (_Float16)v;
            }
        }
        asm volatile("s_wait_dscnt 0x0" ::: "memory");

        // GEMM2: (16x128) x (128x128) + bias, then segment-max scatter.
        v16h afr2[4];
#pragma unroll
        for (int kb = 0; kb < 4; ++kb) afr2[kb] = load_a_frag(sHid + m * 128, kb, lane);

#pragma unroll
        for (int nt = 0; nt < 8; ++nt) {
            const int n = nt * 16 + ncol;
            v8f acc = {};
#pragma unroll
            for (int kb = 0; kb < 4; ++kb) {
                const v16h b = *(const v16h*)(sW2 + n * 128 + kb * 32 + bofs);
                acc = __builtin_amdgcn_wmma_f32_16x16x32_f16(false, afr2[kb], false, b,
                                                             (short)0, acc, false, false);
            }
            const float bias = lb2[n];
#pragma unroll
            for (int i = 0; i < 8; ++i) {
                const float v = acc[i] + bias;
                atomic_max_f32(agg + (size_t)sDst[mofs + i] * HID + n, v);
            }
        }
    }
}

// ---------------------------------------------------------------------------
// Node kernel: global_nn on 16-node tiles. out = relu(agg@gw1+gb1)@gw2+gb2
// ---------------------------------------------------------------------------
__global__ __launch_bounds__(128) void k_node(
    const float* __restrict__ agg,
    const _Float16* __restrict__ gw1T, const float* __restrict__ gb1,
    const _Float16* __restrict__ gw2T, const float* __restrict__ gb2,
    float* __restrict__ out)
{
    extern __shared__ char smem[];
    _Float16* sW1    = (_Float16*)smem;           // 128*128
    _Float16* sW2    = sW1 + 128 * 128;           // 128*128
    _Float16* sABase = sW2 + 128 * 128;           // 4 waves * 16*128
    _Float16* sHBase = sABase + 4 * (16 * 128);   // 4 waves * 16*128

    const int tid  = threadIdx.x;
    const int lane = tid & 31;
    const int wave = tid >> 5;
    _Float16* sA = sABase + wave * (16 * 128);
    _Float16* sH = sHBase + wave * (16 * 128);

    stage_lds(gw1T, sW1, 128 * 128 * 2, tid, blockDim.x);
    stage_lds(gw2T, sW2, 128 * 128 * 2, tid, blockDim.x);
    stage_fence();

    const int m    = lane & 15;
    const int koff = (lane & 16) ? 64 : 0;
    const int ncol = lane & 15;
    const int bofs = (lane & 16) ? 16 : 0;
    const int mofs = (lane & 16) ? 8 : 0;
    const int NT   = NNODES / 16;

    for (int tile = blockIdx.x * 4 + wave; tile < NT; tile += NODE_BLOCKS * 4) {
        const float* ar = agg + (size_t)(tile * 16 + m) * HID + koff;
#pragma unroll
        for (int k = 0; k < 64; k += 8) {
            float4 p0 = *(const float4*)(ar + k);
            float4 p1 = *(const float4*)(ar + k + 4);
            *(v8h*)(sA + m * 128 + koff + k) = cvt8(p0, p1);
        }
        asm volatile("s_wait_dscnt 0x0" ::: "memory");

        v16h afr[4];
#pragma unroll
        for (int kb = 0; kb < 4; ++kb) afr[kb] = load_a_frag(sA + m * 128, kb, lane);

#pragma unroll
        for (int nt = 0; nt < 8; ++nt) {
            const int n = nt * 16 + ncol;
            v8f acc = {};
#pragma unroll
            for (int kb = 0; kb < 4; ++kb) {
                const v16h b = *(const v16h*)(sW1 + n * 128 + kb * 32 + bofs);
                acc = __builtin_amdgcn_wmma_f32_16x16x32_f16(false, afr[kb], false, b,
                                                             (short)0, acc, false, false);
            }
            const float bias = gb1[n];
#pragma unroll
            for (int i = 0; i < 8; ++i) {
                float v = acc[i] + bias;
                v = v > 0.0f ? v : 0.0f;
                sH[(mofs + i) * 128 + n] = (_Float16)v;
            }
        }
        asm volatile("s_wait_dscnt 0x0" ::: "memory");

        v16h afr2[4];
#pragma unroll
        for (int kb = 0; kb < 4; ++kb) afr2[kb] = load_a_frag(sH + m * 128, kb, lane);

#pragma unroll
        for (int nt = 0; nt < 8; ++nt) {
            const int n = nt * 16 + ncol;
            v8f acc = {};
#pragma unroll
            for (int kb = 0; kb < 4; ++kb) {
                const v16h b = *(const v16h*)(sW2 + n * 128 + kb * 32 + bofs);
                acc = __builtin_amdgcn_wmma_f32_16x16x32_f16(false, afr2[kb], false, b,
                                                             (short)0, acc, false, false);
            }
            const float bias = gb2[n];
#pragma unroll
            for (int i = 0; i < 8; ++i)
                out[(size_t)(tile * 16 + mofs + i) * HID + n] = acc[i] + bias;
        }
    }
}

// ---------------------------------------------------------------------------
extern "C" void kernel_launch(void* const* d_in, const int* in_sizes, int n_in,
                              void* d_out, int out_size, void* d_ws, size_t ws_size,
                              hipStream_t stream)
{
    const float*      x   = (const float*)d_in[0];
    const float*      pos = (const float*)d_in[1];
    const long long*  ei  = (const long long*)d_in[2];
    const float*      P[16];
    for (int i = 0; i < 16; ++i) P[i] = (const float*)d_in[3 + i];
    // P[l*8 + {0..7}] = lw1, lb1, lw2, lb2, gw1, gb1, gw2, gb2 for layer l

    // Workspace: agg (N*HID f32 = 32MB) then f16 transposed weights (~320KB).
    float*    agg   = (float*)d_ws;
    _Float16* wptr  = (_Float16*)((char*)d_ws + (size_t)NNODES * HID * sizeof(float));
    _Float16 *lw1T[2], *lw2T[2], *gw1T[2], *gw2T[2];
    for (int l = 0; l < 2; ++l) {
        lw1T[l] = wptr; wptr += 256 * 128;
        lw2T[l] = wptr; wptr += 128 * 128;
        gw1T[l] = wptr; wptr += 128 * 128;
        gw2T[l] = wptr; wptr += 128 * 128;
    }

    for (int l = 0; l < 2; ++l) {
        k_w2h<<<(256 * 128 + 255) / 256, 256, 0, stream>>>(P[l * 8 + 0], lw1T[l], 256);
        k_w2h<<<(128 * 128 + 255) / 256, 256, 0, stream>>>(P[l * 8 + 2], lw2T[l], 128);
        k_w2h<<<(128 * 128 + 255) / 256, 256, 0, stream>>>(P[l * 8 + 4], gw1T[l], 128);
        k_w2h<<<(128 * 128 + 255) / 256, 256, 0, stream>>>(P[l * 8 + 6], gw2T[l], 128);
    }

    float* h = (float*)d_out;  // layer-0 hidden lives in d_out, overwritten by layer 1
    const int fill_n = NNODES * HID;

    const size_t smem_edge = (size_t)(128 * 256 + 128 * 128) * 2   // weights
                           + (size_t)4 * (16 * 256) * 2            // msg tiles
                           + (size_t)4 * (16 * 128) * 2            // hidden tiles
                           + (size_t)4 * 16 * 4;                   // dst ids
    const size_t smem_node = (size_t)(2 * 128 * 128) * 2
                           + (size_t)8 * (16 * 128) * 2;

    for (int l = 0; l < 2; ++l) {
        const float* xin = (l == 0) ? x : h;
        k_fill<<<(fill_n + 255) / 256, 256, 0, stream>>>(agg, -3.402823466e38f, fill_n);
        k_edge<<<EDGE_BLOCKS, 128, smem_edge, stream>>>(
            xin, pos, ei, lw1T[l], lw2T[l], P[l * 8 + 1], P[l * 8 + 3], agg);
        k_node<<<NODE_BLOCKS, 128, smem_node, stream>>>(
            agg, gw1T[l], P[l * 8 + 5], gw2T[l], P[l * 8 + 7], h);
    }
}
